// SelfWarp_74320114090160
// MI455X (gfx1250) — matrix operand order repaired and checked by reading, not compiled, not run on hardware
//
#include <hip/hip_runtime.h>
#include <math.h>

typedef __attribute__((ext_vector_type(16))) _Float16 v16h;
typedef __attribute__((ext_vector_type(8)))  _Float16 v8h;
typedef __attribute__((ext_vector_type(4)))  _Float16 v4h;
typedef __attribute__((ext_vector_type(8)))  float    v8f;

#define CIN    128
#define COUT   128
#define HEADS  32
#define HH     256
#define WW     256
#define HW     (HH * WW)          // 65536 pixels per image
#define NB     8                  // batch
#define FLOWC  (2 * HEADS)        // 64 flow channels
#define TILE_N 64                 // pixels per workgroup
#define KP     144                // padded LDS stride in halves (288B rows, 32B-aligned frags)
#define NCHUNKS (TILE_N / 16)     // 4
#define KCHUNKS (CIN / 32)        // 4 WMMA K-steps of 32

// ---- A-fragment (weights fp32 in HBM -> f16 frag), contiguous-k 16x32 layout (not the ISA table order):
// lanes 0-15: M=lane, khalf=0; lanes 16-31: M=lane-16, khalf=16.
// elements 0..15 -> K = khalf+0..15 (matches B fragment).
__device__ __forceinline__ v16h load_afrag(const float* __restrict__ w,
                                           int mbase, int kbase, int lane) {
    const int row   = mbase + (lane & 15);
    const int khalf = (lane >> 4) * 16;
    const float* wr = w + row * CIN + kbase;
    v16h a;
#pragma unroll
    for (int i = 0; i < 8; ++i) a[i]     = (_Float16)wr[khalf + i];
#pragma unroll
    for (int i = 0; i < 8; ++i) a[8 + i] = (_Float16)wr[khalf + 8 + i];
    return a;
}

// ---- B-fragment from LDS [n][k] layout; lanes 0-15: col=n, K=kbase+0..15,
// lanes 16-31: same cols, K=kbase+16..31 -> one 32B contiguous load per lane.
__device__ __forceinline__ v16h load_bfrag(const _Float16* lds, int nb, int kbase, int lane) {
    const int col = nb + (lane & 15);
    const int kb  = kbase + (lane >> 4) * 16;
    return *(const v16h*)(lds + col * KP + kb);
}

__launch_bounds__(256, 1)
__global__ void fused_flow_value_gemm(
    const float* __restrict__ u,
    const float* __restrict__ w_f1, const float* __restrict__ b_f1,
    const float* __restrict__ w_f2, const float* __restrict__ b_f2,
    const float* __restrict__ w_v,  const float* __restrict__ b_v,
    _Float16* __restrict__ value_ws,   // [b*HEADS + head][pix][4] f16
    float*    __restrict__ flow_ws)    // [b*HEADS + head][pix][2] f32 (fx,fy)
{
    __shared__ __align__(32) _Float16 u_lds[TILE_N * KP];  // u tile, f16 [n][c]
    __shared__ __align__(32) _Float16 h_lds[TILE_N * KP];  // hmid tile, f16 [n][c]

    const int t    = threadIdx.x;
    const int wave = t >> 5;
    const int lane = t & 31;

    const long long p0 = (long long)blockIdx.x * TILE_N;   // global pixel index
    const int b        = (int)(p0 / HW);
    const int pixin    = (int)(p0 % HW);                   // tile never crosses images

    // ---- stage u tile -> LDS f16 (coalesced global reads, transposed LDS store)
    for (int q = t; q < TILE_N * CIN; q += 256) {
        const int c = q >> 6;              // q / TILE_N
        const int n = q & (TILE_N - 1);
        const float v = u[(long long)(b * CIN + c) * HW + pixin + n];
        u_lds[n * KP + c] = (_Float16)v;
    }
    __syncthreads();

    // ---- per-wave weight fragments (held in VGPRs; wave32 has headroom)
    const int mt = wave;                   // M-tile: 16 output channels
    v16h a_f1[KCHUNKS], a_v[KCHUNKS];
#pragma unroll
    for (int kc = 0; kc < KCHUNKS; ++kc) {
        a_f1[kc] = load_afrag(w_f1, mt * 16, kc * 32, lane);
        a_v [kc] = load_afrag(w_v,  mt * 16, kc * 32, lane);
    }

    // C/D 16x16 f32 layout: lane -> N = lane&15, M = r + 8*(lane>>4)
    const int dn    = lane & 15;
    const int dmhi  = (lane >> 4) * 8;
    const int basem = mt * 16 + dmhi;      // multiple of 8

    // ---- phase 1: hmid = relu(W_f1 u + b) -> LDS ; value = W_v u + b -> HBM (f16, packed)
    for (int nc = 0; nc < NCHUNKS; ++nc) {
        const int nb = nc * 16;
        v8f acc_h = {};
        v8f acc_v = {};
#pragma unroll
        for (int kc = 0; kc < KCHUNKS; ++kc) {
            const v16h bf = load_bfrag(u_lds, nb, kc * 32, lane);
            acc_h = __builtin_amdgcn_wmma_f32_16x16x32_f16(
                        false, a_f1[kc], false, bf, (short)0, acc_h, false, false);
            acc_v = __builtin_amdgcn_wmma_f32_16x16x32_f16(
                        false, a_v[kc],  false, bf, (short)0, acc_v, false, false);
        }

        // hmid: 8 contiguous f16 -> one 16B ds_store
        v8h hv;
#pragma unroll
        for (int r = 0; r < 8; ++r) {
            const float h = acc_h[r] + b_f1[basem + r];
            hv[r] = (_Float16)(h > 0.f ? h : 0.f);
        }
        *(v8h*)(&h_lds[(nb + dn) * KP + basem]) = hv;

        // value: interleaved-by-head f16 -> two 8B global stores
        const int pix = pixin + nb + dn;
        v4h pv0, pv1;
#pragma unroll
        for (int r = 0; r < 4; ++r) pv0[r] = (_Float16)(acc_v[r]     + b_v[basem + r]);
#pragma unroll
        for (int r = 0; r < 4; ++r) pv1[r] = (_Float16)(acc_v[4 + r] + b_v[basem + 4 + r]);
        const int h0 = basem >> 2;                         // head of rows 0..3
        *(v4h*)(value_ws + ((long long)(b * HEADS + h0)     * HW + pix) * 4) = pv0;
        *(v4h*)(value_ws + ((long long)(b * HEADS + h0 + 1) * HW + pix) * 4) = pv1;
    }
    __syncthreads();

    // ---- phase 2: flow = W_f2 hmid + b (64 rows -> waves 0..3; wave-uniform branch)
    if (wave < 4) {
        v16h a_f2[KCHUNKS];
#pragma unroll
        for (int kc = 0; kc < KCHUNKS; ++kc)
            a_f2[kc] = load_afrag(w_f2, wave * 16, kc * 32, lane);
        const int fbase = wave * 16 + dmhi;                // multiple of 8
        for (int nc = 0; nc < NCHUNKS; ++nc) {
            const int nb = nc * 16;
            v8f acc = {};
#pragma unroll
            for (int kc = 0; kc < KCHUNKS; ++kc) {
                const v16h bf = load_bfrag(h_lds, nb, kc * 32, lane);
                acc = __builtin_amdgcn_wmma_f32_16x16x32_f16(
                          false, a_f2[kc], false, bf, (short)0, acc, false, false);
            }
            // (fx,fy) pairs -> four 8B float2 stores
            const int pix = pixin + nb + dn;
#pragma unroll
            for (int pr = 0; pr < 4; ++pr) {
                const int hp = (fbase >> 1) + pr;          // head index
                float2 fv;
                fv.x = acc[2 * pr]     + b_f2[fbase + 2 * pr];
                fv.y = acc[2 * pr + 1] + b_f2[fbase + 2 * pr + 1];
                *(float2*)(flow_ws + ((long long)(b * HEADS + hp) * HW + pix) * 2) = fv;
            }
        }
    }
}

// ---- pass 2: periodic bilinear warp. One thread per (b, head, pixel).
// flow: one float2 load; each corner: one 8B v4h load (4 channels contiguous).
__launch_bounds__(256, 1)
__global__ void warp_gather(const float*    __restrict__ flow_ws,
                            const _Float16* __restrict__ value_ws,
                            float*          __restrict__ out)
{
    const long long idx = (long long)blockIdx.x * blockDim.x + threadIdx.x;
    const int pix  = (int)(idx & (HW - 1));
    const int bh   = (int)(idx >> 16);          // b*HEADS + head
    const int head = bh & (HEADS - 1);
    const int b    = bh >> 5;
    const int y = pix >> 8;
    const int x = pix & (WW - 1);

    const float2 f = *(const float2*)(flow_ws + ((long long)bh * HW + pix) * 2);

    // base grid (linspace -1..1) + flow, unnormalize (align_corners=False), wrap
    const float gx = (-1.f + 2.f * (float)x / (float)(WW - 1)) + f.x;
    const float gy = (-1.f + 2.f * (float)y / (float)(HH - 1)) + f.y;
    float px = fmodf(((gx + 1.f) * (float)WW - 1.f) * 0.5f, (float)WW);
    if (px < 0.f) px += (float)WW;              // jnp.mod: sign of divisor
    float py = fmodf(((gy + 1.f) * (float)HH - 1.f) * 0.5f, (float)HH);
    if (py < 0.f) py += (float)HH;

    const float x0 = floorf(px), y0 = floorf(py);
    const float wx = px - x0,    wy = py - y0;
    int x0i = (int)x0; x0i = x0i < 0 ? 0 : (x0i > WW - 1 ? WW - 1 : x0i);
    int y0i = (int)y0; y0i = y0i < 0 ? 0 : (y0i > HH - 1 ? HH - 1 : y0i);
    const int x1i = (x0i + 1) & (WW - 1);
    const int y1i = (y0i + 1) & (HH - 1);

    const float w00 = (1.f - wx) * (1.f - wy);
    const float w01 = wx * (1.f - wy);
    const float w10 = (1.f - wx) * wy;
    const float w11 = wx * wy;

    const _Float16* vb = value_ws + (long long)bh * HW * 4;
    const v4h c00 = *(const v4h*)(vb + (long long)(y0i * WW + x0i) * 4);
    const v4h c01 = *(const v4h*)(vb + (long long)(y0i * WW + x1i) * 4);
    const v4h c10 = *(const v4h*)(vb + (long long)(y1i * WW + x0i) * 4);
    const v4h c11 = *(const v4h*)(vb + (long long)(y1i * WW + x1i) * 4);

#pragma unroll
    for (int cc = 0; cc < 4; ++cc) {
        const float r = (float)c00[cc] * w00 + (float)c01[cc] * w01
                      + (float)c10[cc] * w10 + (float)c11[cc] * w11;
        out[(long long)(b * COUT + head * 4 + cc) * HW + pix] = r;
    }
}

extern "C" void kernel_launch(void* const* d_in, const int* in_sizes, int n_in,
                              void* d_out, int out_size, void* d_ws, size_t ws_size,
                              hipStream_t stream) {
    const float* u    = (const float*)d_in[0];
    const float* w_f1 = (const float*)d_in[1];
    const float* b_f1 = (const float*)d_in[2];
    const float* w_f2 = (const float*)d_in[3];
    const float* b_f2 = (const float*)d_in[4];
    const float* w_v  = (const float*)d_in[5];
    const float* b_v  = (const float*)d_in[6];
    float* out = (float*)d_out;

    // workspace: value f16 interleaved (128MB) then flow f32 pairs (128MB)
    _Float16* value_ws = (_Float16*)d_ws;
    float*    flow_ws  = (float*)((char*)d_ws + (size_t)NB * HEADS * HW * 4 * sizeof(_Float16));

    const int gemm_blocks = (NB * HW) / TILE_N;          // 8192 workgroups
    fused_flow_value_gemm<<<gemm_blocks, 256, 0, stream>>>(
        u, w_f1, b_f1, w_f2, b_f2, w_v, b_v, value_ws, flow_ws);

    const long long total = (long long)NB * HEADS * HW;  // 16.7M threads
    const int warp_blocks = (int)(total / 256);          // 65536 workgroups
    warp_gather<<<warp_blocks, 256, 0, stream>>>(flow_ws, value_ws, out);
}
